// CausalSelfAttention_68805376082019
// MI455X (gfx1250) — compile-verified, matrix operand order checked by reading, not run on hardware
//
#include <hip/hip_runtime.h>
#include <hip/hip_bf16.h>
#include <stdint.h>

// ---------------------------------------------------------------------------
// CDNA5 (gfx1250) causal self-attention, bf16 WMMA with f32 accumulation.
// ---------------------------------------------------------------------------

#define N_EMBD 1024
#define N_HEADS 16
#define HEAD_DIM 64
#define BATCH 4
#define SEQ 2048
#define MROWS (BATCH * SEQ)  // 8192

typedef __bf16 bf16_t;
typedef __attribute__((ext_vector_type(16))) __bf16 v16bf;
typedef __attribute__((ext_vector_type(8)))  __bf16 v8bf;
typedef __attribute__((ext_vector_type(8)))  float  v8f;

__device__ inline uint16_t f2bf(float f) {
    union { float f; uint32_t u; } c; c.f = f;
    uint32_t u = c.u;
    uint32_t r = (u + 0x7FFFu + ((u >> 16) & 1u)) >> 16;
    return (uint16_t)r;
}

__device__ inline v8f zero8() {
    v8f z;
#pragma unroll
    for (int i = 0; i < 8; i++) z[i] = 0.0f;
    return z;
}

// Load a 16-element bf16 fragment following the CDNA5 16-bit operand layout:
// elements 0..7 come from base[0..7], elements 8..15 from base[16..23].
__device__ inline v16bf load_frag(const uint16_t* base) {
    v8bf lo = *(const v8bf*)(base);
    v8bf hi = *(const v8bf*)(base + 16);
    v16bf r;
#pragma unroll
    for (int i = 0; i < 8; i++) { r[i] = lo[i]; r[i + 8] = hi[i]; }
    return r;
}

__device__ inline v8f wmma_bf16(v16bf a, v16bf b, v8f c) {
    // D(f32 16x16) = A(bf16 16x32) * B(bf16 32x16) + C
    return __builtin_amdgcn_wmma_f32_16x16x32_bf16(
        /*neg_a=*/false, a, /*neg_b=*/false, b,
        /*c_mod=*/(short)0, c, /*reuse_a=*/false, /*reuse_b=*/false);
}

// ---------------------------------------------------------------------------
// Kernel 1: f32 -> bf16 cast (grid-stride)
// ---------------------------------------------------------------------------
__global__ __launch_bounds__(256) void cast_f32_bf16(const float* __restrict__ in,
                                                     uint16_t* __restrict__ out,
                                                     size_t n) {
    size_t i = (size_t)blockIdx.x * blockDim.x + threadIdx.x;
    size_t stride = (size_t)gridDim.x * blockDim.x;
    for (; i < n; i += stride) out[i] = f2bf(in[i]);
}

// ---------------------------------------------------------------------------
// Kernel 2: C = A[M,K] * W[N,K]^T + bias  (bf16 in, f32 accumulate)
// One wave computes a 16(M) x 64(N) tile: A-fragment reused for 4 WMMAs.
// mode 0: store bf16 head-major   [B, H, T, hd]        (Q, K)
// mode 1: store bf16 head-major-T [B, H, hd, T]        (V, pre-transposed)
// mode 2: store f32 row-major     [M, N]               (final projection)
// ---------------------------------------------------------------------------
__global__ __launch_bounds__(256) void gemm_bf16_kernel(
    const uint16_t* __restrict__ A, const uint16_t* __restrict__ W,
    const float* __restrict__ bias, void* __restrict__ outp,
    int M, int N, int K, int mode) {
    const int wid  = blockIdx.x * (blockDim.x >> 5) + (threadIdx.x >> 5);
    const int lane = threadIdx.x & 31;
    const int tn64 = N >> 6;
    const int tm = wid / tn64;
    const int tn = wid - tm * tn64;
    if (tm >= (M >> 4)) return;  // wave-uniform guard

    const int m0 = tm << 4, n0 = tn << 6;
    const int half = lane >> 4;       // 0: lanes 0-15, 1: lanes 16-31
    const int l16  = lane & 15;
    const int koff = half << 3;       // K sub-offset per lane half

    const uint16_t* arow = A + (size_t)(m0 + l16) * K;

    v8f acc[4];
#pragma unroll
    for (int j = 0; j < 4; j++) acc[j] = zero8();

    for (int kk = 0; kk < K; kk += 32) {
        if (kk + 32 < K) {
            __builtin_prefetch((const void*)(arow + kk + 32), 0, 0);
        }
        v16bf a = load_frag(arow + kk + koff);
#pragma unroll
        for (int j = 0; j < 4; j++) {
            const uint16_t* wr = W + (size_t)(n0 + j * 16 + l16) * K + kk + koff;
            v16bf b = load_frag(wr);
            acc[j] = wmma_bf16(a, b, acc[j]);
        }
    }

    // C layout: lane holds col = n0 + j*16 + l16; VGPR r holds row m0 + r + 8*half
    if (mode == 2) {
        float* out = (float*)outp;
#pragma unroll
        for (int j = 0; j < 4; j++) {
            const int col = n0 + j * 16 + l16;
            const float bv = bias[col];
#pragma unroll
            for (int r = 0; r < 8; r++) {
                const int row = m0 + r + (half << 3);
                out[(size_t)row * N + col] = acc[j][r] + bv;
            }
        }
    } else {
        uint16_t* out = (uint16_t*)outp;
#pragma unroll
        for (int j = 0; j < 4; j++) {
            const int col = n0 + j * 16 + l16;
            const float bv = bias[col];
            const int h = col >> 6, d = col & 63;
#pragma unroll
            for (int r = 0; r < 8; r++) {
                const int row = m0 + r + (half << 3);
                const int b = row >> 11, t = row & 2047;
                size_t idx;
                if (mode == 0)
                    idx = (((size_t)(b * N_HEADS + h) * SEQ) + t) * HEAD_DIM + d;
                else
                    idx = (((size_t)(b * N_HEADS + h) * HEAD_DIM) + d) * SEQ + t;
                out[idx] = f2bf(acc[j][r] + bv);
            }
        }
    }
}

// ---------------------------------------------------------------------------
// Kernel 3: flash-style causal attention.
// One wave = 16 queries of one (b,h). Streams keys in 32-wide tiles.
//   S = Q K^T  : 4 x v_wmma_f32_16x16x32_bf16 per tile
//   online softmax in f32 with __shfl_xor row reductions (wave32)
//   P (C-layout f32) -> bf16 A-layout via LDS bounce + s_wait_dscnt
//   O += P V   : 4 x v_wmma (V pre-transposed [hd][T] -> contiguous B frags)
// ---------------------------------------------------------------------------
__global__ __launch_bounds__(256) void attn_kernel(
    const uint16_t* __restrict__ Qh, const uint16_t* __restrict__ Kh,
    const uint16_t* __restrict__ Vt, uint16_t* __restrict__ Yb) {
    __shared__ uint16_t pbuf[8 * 16 * 32];  // per-wave 16x32 bf16 P tile

    const int wid  = blockIdx.x * (blockDim.x >> 5) + (threadIdx.x >> 5);
    const int lane = threadIdx.x & 31;
    const int bh = wid >> 7;        // 0..63  (b*16 + h)
    const int qt = wid & 127;       // query tile
    const int q0 = qt << 4;
    const int half = lane >> 4;
    const int l16  = lane & 15;
    const int koff = half << 3;

    const uint16_t* Qb = Qh + (size_t)bh * SEQ * HEAD_DIM;
    const uint16_t* Kb = Kh + (size_t)bh * SEQ * HEAD_DIM;
    const uint16_t* Vb = Vt + (size_t)bh * HEAD_DIM * SEQ;
    uint16_t* pl = &pbuf[(threadIdx.x >> 5) * (16 * 32)];

    // Q A-fragments for the two 32-wide d-chunks (lane = query row l16)
    const int qrow = q0 + l16;
    const v16bf qf0 = load_frag(Qb + (size_t)qrow * HEAD_DIM + 0  + koff);
    const v16bf qf1 = load_frag(Qb + (size_t)qrow * HEAD_DIM + 32 + koff);

    float mrow[8], lrow[8];
#pragma unroll
    for (int r = 0; r < 8; r++) { mrow[r] = -3.0e38f; lrow[r] = 0.0f; }
    v8f o[4];
#pragma unroll
    for (int j = 0; j < 4; j++) o[j] = zero8();

    const float scale = 0.125f;  // 1/sqrt(64)

    for (int k0 = 0; k0 < q0 + 16; k0 += 32) {
        // ---- S = Q K^T for keys [k0, k0+32) : two 16-col tiles -------------
        v8f s0 = zero8(), s1 = zero8();
        {
            const uint16_t* kb0 = Kb + (size_t)(k0 + l16) * HEAD_DIM;
            s0 = wmma_bf16(qf0, load_frag(kb0 + 0  + koff), s0);
            s0 = wmma_bf16(qf1, load_frag(kb0 + 32 + koff), s0);
            const uint16_t* kb1 = Kb + (size_t)(k0 + 16 + l16) * HEAD_DIM;
            s1 = wmma_bf16(qf0, load_frag(kb1 + 0  + koff), s1);
            s1 = wmma_bf16(qf1, load_frag(kb1 + 32 + koff), s1);
        }

        // ---- scale, causal mask, online softmax ---------------------------
        float p0[8], p1[8];
#pragma unroll
        for (int r = 0; r < 8; r++) {
            const int rowg = q0 + r + (half << 3);
            float v0 = s0[r] * scale;
            float v1 = s1[r] * scale;
            if (k0 + l16 > rowg)      v0 = -3.0e38f;
            if (k0 + 16 + l16 > rowg) v1 = -3.0e38f;
            float rm = fmaxf(v0, v1);
            rm = fmaxf(rm, __shfl_xor(rm, 1, 32));
            rm = fmaxf(rm, __shfl_xor(rm, 2, 32));
            rm = fmaxf(rm, __shfl_xor(rm, 4, 32));
            rm = fmaxf(rm, __shfl_xor(rm, 8, 32));
            const float mn = fmaxf(mrow[r], rm);
            const float corr = __expf(mrow[r] - mn);
            mrow[r] = mn;
            const float e0 = __expf(v0 - mn);
            const float e1 = __expf(v1 - mn);
            float ps = e0 + e1;
            ps += __shfl_xor(ps, 1, 32);
            ps += __shfl_xor(ps, 2, 32);
            ps += __shfl_xor(ps, 4, 32);
            ps += __shfl_xor(ps, 8, 32);
            lrow[r] = lrow[r] * corr + ps;
            p0[r] = e0; p1[r] = e1;
#pragma unroll
            for (int j = 0; j < 4; j++) o[j][r] *= corr;
        }

        // ---- P: C-layout -> A-layout via LDS ------------------------------
#pragma unroll
        for (int r = 0; r < 8; r++) {
            const int rl = r + (half << 3);
            pl[rl * 32 + l16]      = f2bf(p0[r]);
            pl[rl * 32 + 16 + l16] = f2bf(p1[r]);
        }
        asm volatile("s_wait_dscnt 0" ::: "memory");
        const v16bf pf = load_frag(pl + l16 * 32 + koff);

        // ---- O += P V  (V transposed: row d contiguous along keys) --------
#pragma unroll
        for (int j = 0; j < 4; j++) {
            const int d = j * 16 + l16;
            const v16bf vf = load_frag(Vb + (size_t)d * SEQ + k0 + koff);
            o[j] = wmma_bf16(pf, vf, o[j]);
        }
    }

    // ---- normalize and store y as bf16 [B,T,C] ----------------------------
    const int b = bh >> 4, h = bh & 15;
#pragma unroll
    for (int r = 0; r < 8; r++) {
        const float inv = 1.0f / lrow[r];
        const int t = q0 + r + (half << 3);
        const size_t rowoff = ((size_t)b * SEQ + t) * N_EMBD + h * HEAD_DIM;
#pragma unroll
        for (int j = 0; j < 4; j++)
            Yb[rowoff + j * 16 + l16] = f2bf(o[j][r] * inv);
    }
}

// ---------------------------------------------------------------------------
// Launcher
// ---------------------------------------------------------------------------
extern "C" void kernel_launch(void* const* d_in, const int* in_sizes, int n_in,
                              void* d_out, int out_size, void* d_ws, size_t ws_size,
                              hipStream_t stream) {
    (void)in_sizes; (void)n_in; (void)out_size; (void)ws_size;

    const float* x  = (const float*)d_in[0];
    const float* Wq = (const float*)d_in[1];
    const float* bq = (const float*)d_in[2];
    const float* Wk = (const float*)d_in[3];
    const float* bk = (const float*)d_in[4];
    const float* Wv = (const float*)d_in[5];
    const float* bv = (const float*)d_in[6];
    const float* Wc = (const float*)d_in[7];
    const float* bc = (const float*)d_in[8];

    char* ws = (char*)d_ws;
    size_t off = 0;
    const size_t XB   = (size_t)MROWS * N_EMBD;           // 8.4M elems
    const size_t WB   = (size_t)N_EMBD * N_EMBD;          // 1M elems
    uint16_t* Xb  = (uint16_t*)(ws + off); off += XB * 2;
    uint16_t* Wqb = (uint16_t*)(ws + off); off += WB * 2;
    uint16_t* Wkb = (uint16_t*)(ws + off); off += WB * 2;
    uint16_t* Wvb = (uint16_t*)(ws + off); off += WB * 2;
    uint16_t* Wcb = (uint16_t*)(ws + off); off += WB * 2;
    uint16_t* Qhb = (uint16_t*)(ws + off); off += XB * 2;
    uint16_t* Khb = (uint16_t*)(ws + off); off += XB * 2;
    uint16_t* Vtb = (uint16_t*)(ws + off); off += XB * 2;
    uint16_t* Yb  = (uint16_t*)(ws + off); off += XB * 2;

    // 1) casts to bf16
    cast_f32_bf16<<<4096, 256, 0, stream>>>(x,  Xb,  XB);
    cast_f32_bf16<<<1024, 256, 0, stream>>>(Wq, Wqb, WB);
    cast_f32_bf16<<<1024, 256, 0, stream>>>(Wk, Wkb, WB);
    cast_f32_bf16<<<1024, 256, 0, stream>>>(Wv, Wvb, WB);
    cast_f32_bf16<<<1024, 256, 0, stream>>>(Wc, Wcb, WB);

    // 2) QKV projections: (8192/16)*(1024/64) = 8192 waves -> 1024 blocks x 8 waves
    const int gemm_blocks = (MROWS / 16) * (N_EMBD / 64) / 8;
    gemm_bf16_kernel<<<gemm_blocks, 256, 0, stream>>>(Xb, Wqb, bq, (void*)Qhb,
                                                      MROWS, N_EMBD, N_EMBD, 0);
    gemm_bf16_kernel<<<gemm_blocks, 256, 0, stream>>>(Xb, Wkb, bk, (void*)Khb,
                                                      MROWS, N_EMBD, N_EMBD, 0);
    gemm_bf16_kernel<<<gemm_blocks, 256, 0, stream>>>(Xb, Wvb, bv, (void*)Vtb,
                                                      MROWS, N_EMBD, N_EMBD, 1);

    // 3) attention: 64 (b,h) * 128 query tiles = 8192 waves -> 1024 blocks
    attn_kernel<<<1024, 256, 0, stream>>>(Qhb, Khb, Vtb, Yb);

    // 4) output projection -> f32 d_out
    gemm_bf16_kernel<<<gemm_blocks, 256, 0, stream>>>(Yb, Wcb, bc, d_out,
                                                      MROWS, N_EMBD, N_EMBD, 2);
}
